// ExpertParallelMoE_89945205113201
// MI455X (gfx1250) — matrix operand order repaired and checked by reading, not compiled, not run on hardware
//
#include <hip/hip_runtime.h>
#include <hip/hip_bf16.h>

// ---- problem constants (from reference) ----
#define TT  2048
#define DD  1024
#define DHH 256
#define EE  64      // routed experts
#define EAA 66      // routed + 2 shared (unified)
#define KRR 6       // routed experts per token
#define KSS 2       // shared experts
#define MT  32      // tokens per expert tile (2 WMMA m-tiles)

typedef __attribute__((ext_vector_type(16))) __bf16 v16bf;
typedef __attribute__((ext_vector_type(8)))  __bf16 v8bf;
typedef __attribute__((ext_vector_type(4)))  __bf16 v4bf;
typedef __attribute__((ext_vector_type(8)))  float  v8f;
typedef __attribute__((ext_vector_type(4)))  float  v4f;

// Build a 16x32 bf16 A-fragment for one lane from a k-contiguous row.
// Contiguous layout: lanes 0-15 hold K {k0..k0+15} for M=lane,
// lanes 16-31 hold K {k0+16..k0+31} for M=lane-16.
__device__ __forceinline__ v16bf load_frag_row(const __bf16* rowp, int k0, int hfl) {
    v8bf lo = *(const v8bf*)(rowp + k0 + hfl * 16);
    v8bf hi = *(const v8bf*)(rowp + k0 + hfl * 16 + 8);
    return __builtin_shufflevector(lo, hi,
        0,1,2,3,4,5,6,7,8,9,10,11,12,13,14,15);
}

// ---------------------------------------------------------------------------
// Transpose + convert f32 [R][C] -> bf16 [C][R], per expert (z).
// z < EE pulls from routed weights, z >= EE from shared weights.
// ---------------------------------------------------------------------------
__global__ __launch_bounds__(256) void transpose_cvt_kernel(
    const float* __restrict__ Wr, const float* __restrict__ Ws,
    __bf16* __restrict__ dst, int R, int C)
{
    __shared__ float tile[32][33];
    const int z = blockIdx.z;
    const size_t mat = (size_t)R * C;
    const float* src = (z < EE) ? (Wr + (size_t)z * mat)
                                : (Ws + (size_t)(z - EE) * mat);
    __bf16* o = dst + (size_t)z * mat;
    const int c0 = blockIdx.x * 32, r0 = blockIdx.y * 32;
    const int tx = threadIdx.x, ty = threadIdx.y;
    #pragma unroll
    for (int i = ty; i < 32; i += 8)
        tile[i][tx] = src[(size_t)(r0 + i) * C + c0 + tx];
    __syncthreads();
    #pragma unroll
    for (int i = ty; i < 32; i += 8)
        o[(size_t)(c0 + i) * R + r0 + tx] = (__bf16)tile[tx][i];
}

// ---------------------------------------------------------------------------
// Gating: one wave32 per token. Lane l owns logits for experts l and l+32.
// Top-6 via iterative wave argmax (low-index tie-break, matches lax.top_k),
// softmax over selected logits, scatter (token, score) into per-expert lists.
// ---------------------------------------------------------------------------
__global__ __launch_bounds__(32) void gate_kernel(
    const float* __restrict__ u, const float* __restrict__ Wg,
    int* __restrict__ counts, int* __restrict__ tok_list,
    float* __restrict__ score_list)
{
    const int t = blockIdx.x;
    const int lane = threadIdx.x;
    const float* ur = u + (size_t)t * DD;

    float acc0 = 0.f, acc1 = 0.f;
    for (int d0 = 0; d0 < DD; d0 += 32) {
        float uv = ur[d0 + lane];
        #pragma unroll
        for (int j = 0; j < 32; ++j) {
            float us = __shfl(uv, j, 32);
            const float* wr = Wg + (size_t)(d0 + j) * EE;
            acc0 += us * wr[lane];
            acc1 += us * wr[lane + 32];
        }
    }

    float v0 = acc0, v1 = acc1;
    float tv[KRR]; int te[KRR];
    #pragma unroll
    for (int k = 0; k < KRR; ++k) {
        float bv = v0; int be = lane;
        if (v1 > bv) { bv = v1; be = lane + 32; }
        #pragma unroll
        for (int off = 16; off > 0; off >>= 1) {
            float ov = __shfl_xor(bv, off, 32);
            int   oe = __shfl_xor(be, off, 32);
            if (ov > bv || (ov == bv && oe < be)) { bv = ov; be = oe; }
        }
        tv[k] = bv; te[k] = be;
        if (lane == (be & 31)) {
            if (be < 32) v0 = -3.4e38f; else v1 = -3.4e38f;
        }
    }
    float s = 0.f;
    #pragma unroll
    for (int k = 0; k < KRR; ++k) s += __expf(tv[k] - tv[0]);
    if (lane < KRR) {
        float w = __expf(tv[lane] - tv[0]) / s;
        int ex = te[lane];
        int slot = atomicAdd(&counts[ex], 1);
        tok_list[(size_t)ex * TT + slot]   = t;
        score_list[(size_t)ex * TT + slot] = w;
    }
}

// ---------------------------------------------------------------------------
// Fused FC1+FC2 expert kernel. Block = 8 waves, one (expert, 32-token tile).
// out[tok] += w * (relu(u[tok] @ W1_e + b1_e) @ W2_e + b2_e)
// Shared experts (e >= EE): every token, w = 1/KSS.
// 32-token M-tile => every weight B-fragment feeds 2 WMMAs (halves L2 weight
// traffic vs a 16-row tile; weights are bf16 and fully L2-resident).
// ---------------------------------------------------------------------------
__global__ __launch_bounds__(256) void moe_expert_kernel(
    const float* __restrict__ u,
    const float* __restrict__ br1, const float* __restrict__ bs1,
    const float* __restrict__ br2, const float* __restrict__ bs2,
    const __bf16* __restrict__ W1t,   // [EAA][DHH][DD]  (B for FC1: [N][K])
    const __bf16* __restrict__ W2t,   // [EAA][DD][DHH]  (B for FC2: [N][K])
    const int* __restrict__ tok_list, const float* __restrict__ score_list,
    const int* __restrict__ counts, float* __restrict__ out)
{
    const int e    = blockIdx.y;
    const int tile = blockIdx.x;
    const int cnt  = (e < EE) ? counts[e] : TT;
    if (tile * MT >= cnt) return;          // block-uniform exit (EXEC stays full)

    __shared__ __bf16 sU[MT][DD];          // 64 KB: gathered token rows, bf16
    __shared__ __bf16 sGH[MT][DHH];        // 16 KB: gated hidden (A for FC2)
    __shared__ int    sTok[MT];
    __shared__ float  sW[MT];

    const int tid  = threadIdx.x;
    const int lane = tid & 31;
    const int wave = tid >> 5;
    const int mrow = lane & 15;
    const int hfl  = lane >> 4;            // which 16-lane half

    if (tid < MT) {
        int g = tile * MT + tid;
        int tok = 0; float w = 0.f;
        if (g < cnt) {
            if (e < EE) { tok = tok_list[(size_t)e * TT + g];
                          w   = score_list[(size_t)e * TT + g]; }
            else        { tok = g; w = 1.0f / KSS; }
        }
        sTok[tid] = tok; sW[tid] = w;
    }
    __syncthreads();

    // stage MT gathered token rows as bf16 (float4 loads, 8B LDS stores)
    for (int idx = tid; idx < MT * (DD / 4); idx += 256) {
        int m  = idx >> 8;                 // DD/4 = 256 float4 per row
        int c4 = idx & 255;
        v4f v = *(const v4f*)(u + (size_t)sTok[m] * DD + (size_t)c4 * 4);
        v4bf b;
        b[0] = (__bf16)v[0]; b[1] = (__bf16)v[1];
        b[2] = (__bf16)v[2]; b[3] = (__bf16)v[3];
        *(v4bf*)&sU[m][c4 * 4] = b;
    }
    __syncthreads();

    const float*  b1 = (e < EE) ? (br1 + (size_t)e * DHH) : (bs1 + (size_t)(e - EE) * DHH);
    const float*  b2 = (e < EE) ? (br2 + (size_t)e * DD)  : (bs2 + (size_t)(e - EE) * DD);
    const __bf16* w1 = W1t + (size_t)e * DHH * DD;
    const __bf16* w2 = W2t + (size_t)e * DD * DHH;

    // ========== FC1: h[32 x 256]; each wave: 2 n-tiles x 2 m-tiles =========
    v8f c[2][2];                           // [n-tile][m-tile]
    #pragma unroll
    for (int i = 0; i < 2; ++i)
        #pragma unroll
        for (int m = 0; m < 2; ++m)
            #pragma unroll
            for (int j = 0; j < 8; ++j) c[i][m][j] = 0.f;

    const int n0 = wave * 32 + mrow;       // this lane's output column, tile A
    const int n1 = n0 + 16;                // tile B
    const __bf16* bp0 = w1 + (size_t)n0 * DD;
    const __bf16* bp1 = w1 + (size_t)n1 * DD;

    for (int k0 = 0; k0 < DD; k0 += 32) {
        v16bf a0 = load_frag_row(&sU[mrow][0],      k0, hfl);
        v16bf a1 = load_frag_row(&sU[mrow + 16][0], k0, hfl);
        int kb = k0 + hfl * 16;
        v16bf b0v = *(const v16bf*)(bp0 + kb);
        v16bf b1v = *(const v16bf*)(bp1 + kb);
        c[0][0] = __builtin_amdgcn_wmma_f32_16x16x32_bf16(false, a0, false, b0v,
                                                  (short)0, c[0][0], false, false);
        c[0][1] = __builtin_amdgcn_wmma_f32_16x16x32_bf16(false, a1, false, b0v,
                                                  (short)0, c[0][1], false, false);
        c[1][0] = __builtin_amdgcn_wmma_f32_16x16x32_bf16(false, a0, false, b1v,
                                                  (short)0, c[1][0], false, false);
        c[1][1] = __builtin_amdgcn_wmma_f32_16x16x32_bf16(false, a1, false, b1v,
                                                  (short)0, c[1][1], false, false);
    }
    // epilogue: relu(x + b1) * gate, store gh to LDS (A matrix for FC2)
    {
        float bA = b1[n0], bB = b1[n1];
        #pragma unroll
        for (int mt = 0; mt < 2; ++mt) {
            #pragma unroll
            for (int r = 0; r < 8; ++r) {
                int m = mt * 16 + r + hfl * 8;   // C layout: VGPR r -> M=r (+8)
                float wv = sW[m];
                float hA = c[0][mt][r] + bA; hA = hA > 0.f ? hA : 0.f;
                float hB = c[1][mt][r] + bB; hB = hB > 0.f ? hB : 0.f;
                sGH[m][n0] = (__bf16)(hA * wv);
                sGH[m][n1] = (__bf16)(hB * wv);
            }
        }
    }
    __syncthreads();

    // ========== FC2: y[32 x 1024]; 2 passes of 4 n-tiles x 2 m-tiles =======
    #pragma unroll
    for (int pass = 0; pass < 2; ++pass) {
        v8f acc[4][2];                     // [n-tile][m-tile]
        #pragma unroll
        for (int i = 0; i < 4; ++i)
            #pragma unroll
            for (int m = 0; m < 2; ++m)
                #pragma unroll
                for (int j = 0; j < 8; ++j) acc[i][m][j] = 0.f;

        for (int k0 = 0; k0 < DHH; k0 += 32) {
            v16bf a0 = load_frag_row(&sGH[mrow][0],      k0, hfl);
            v16bf a1 = load_frag_row(&sGH[mrow + 16][0], k0, hfl);
            int kb = k0 + hfl * 16;
            #pragma unroll
            for (int i = 0; i < 4; ++i) {
                int n = (wave * 8 + pass * 4 + i) * 16 + mrow;
                v16bf b = *(const v16bf*)(w2 + (size_t)n * DHH + kb);
                acc[i][0] = __builtin_amdgcn_wmma_f32_16x16x32_bf16(false, a0, false, b,
                                                    (short)0, acc[i][0], false, false);
                acc[i][1] = __builtin_amdgcn_wmma_f32_16x16x32_bf16(false, a1, false, b,
                                                    (short)0, acc[i][1], false, false);
            }
        }
        // epilogue: + w*b2, scatter-accumulate into out (skip pad rows)
        #pragma unroll
        for (int i = 0; i < 4; ++i) {
            int n = (wave * 8 + pass * 4 + i) * 16 + mrow;
            float b2n = b2[n];
            #pragma unroll
            for (int mt = 0; mt < 2; ++mt) {
                #pragma unroll
                for (int r = 0; r < 8; ++r) {
                    int m = mt * 16 + r + hfl * 8;
                    int g = tile * MT + m;
                    if (g < cnt) {
                        float y = acc[i][mt][r] + sW[m] * b2n;
                        unsafeAtomicAdd(out + (size_t)sTok[m] * DD + n, y);
                    }
                }
            }
        }
    }
}

// ---------------------------------------------------------------------------
// Host launcher. Workspace layout (bytes, 256-aligned blocks):
//   [0)           counts      : 64 ints (padded to 1024)
//   [1024)        tok_list    : EE*TT ints        (512 KB)
//   [525312)      score_list  : EE*TT floats      (512 KB)
//   [1049600)     W1t bf16    : EAA*DHH*DD        (~33 MB)
//   [35652608)    W2t bf16    : EAA*DD*DHH        (~33 MB)
// Total ~70.3 MB.
// ---------------------------------------------------------------------------
extern "C" void kernel_launch(void* const* d_in, const int* in_sizes, int n_in,
                              void* d_out, int out_size, void* d_ws, size_t ws_size,
                              hipStream_t stream) {
    const float* u   = (const float*)d_in[0];
    const float* Wg  = (const float*)d_in[1];
    const float* Ws1 = (const float*)d_in[2];
    const float* bs1 = (const float*)d_in[3];
    const float* Ws2 = (const float*)d_in[4];
    const float* bs2 = (const float*)d_in[5];
    const float* Wr1 = (const float*)d_in[6];
    const float* br1 = (const float*)d_in[7];
    const float* Wr2 = (const float*)d_in[8];
    const float* br2 = (const float*)d_in[9];
    float* out = (float*)d_out;

    char* ws = (char*)d_ws;
    int*    counts = (int*)ws;
    int*    tok    = (int*)(ws + 1024);
    float*  score  = (float*)(ws + 1024 + (size_t)EE * TT * 4);
    __bf16* W1t    = (__bf16*)(ws + 1049600);
    __bf16* W2t    = (__bf16*)(ws + 35652608);

    hipMemsetAsync(counts, 0, 64 * sizeof(int), stream);
    hipMemcpyAsync(out, u, (size_t)TT * DD * sizeof(float),
                   hipMemcpyDeviceToDevice, stream);   // residual +u

    dim3 tb(32, 8);
    // W1 [D][DH] -> [DH][D] bf16 ; W2 [DH][D] -> [D][DH] bf16
    transpose_cvt_kernel<<<dim3(DHH / 32, DD / 32, EAA), tb, 0, stream>>>(
        Wr1, Ws1, W1t, DD, DHH);
    transpose_cvt_kernel<<<dim3(DD / 32, DHH / 32, EAA), tb, 0, stream>>>(
        Wr2, Ws2, W2t, DHH, DD);

    gate_kernel<<<TT, 32, 0, stream>>>(u, Wg, counts, tok, score);

    moe_expert_kernel<<<dim3(TT / MT, EAA), 256, 0, stream>>>(
        u, br1, bs1, br2, bs2, W1t, W2t, tok, score, counts, out);
}
